// ElectronicEmbedding_17179869975
// MI455X (gfx1250) — matrix operand order repaired and checked by reading, not compiled, not run on hardware
//
#include <hip/hip_runtime.h>
#include <hip/hip_bf16.h>
#include <math.h>

typedef __attribute__((ext_vector_type(16))) _Float16 v16h;
typedef __attribute__((ext_vector_type(8)))  _Float16 v8h;
typedef __attribute__((ext_vector_type(4)))  _Float16 v4h;
typedef __attribute__((ext_vector_type(8)))  float    v8f;

#define D128 128
#define TW   136   // f16 tile row stride in halves (128 + 8 pad -> bank-conflict free)
#define XW   132   // f32 tile row stride in floats (128 + 4 pad)
#define EPS  1e-8f

// fast silu: result feeds f16, so v_rcp_f32 (~1 ulp) is more than enough —
// avoids the v_div_scale/v_div_fmas IEEE divide sequence seen in round-1 asm.
__device__ __forceinline__ float silu_f(float x) {
    return x * __builtin_amdgcn_rcpf(1.0f + __expf(-x));
}

// build a v16h fragment from two 16-byte LDS chunks
__device__ __forceinline__ v16h ld_frag2(const _Float16* p0, const _Float16* p1) {
    v8h a = *(const v8h*)p0;
    v8h b = *(const v8h*)p1;
    v16h r;
#pragma unroll
    for (int i = 0; i < 8; ++i) { r[i] = a[i]; r[8 + i] = b[i]; }
    return r;
}

// ---------------------------------------------------------------- zero anorm
__global__ void zero_f32_kernel(float* p, int n) {
    int i = blockIdx.x * blockDim.x + threadIdx.x;
    if (i < n) p[i] = 0.0f;
}

// ------------------------------------------------- per-molecule prep (B blocks)
// k = (e/enorm)@Wk.T ; v = e@Wv.T ; kq = Wq^T k ; c = k.bq
__global__ void mol_prep_kernel(const float* __restrict__ Q,
                                const float* __restrict__ Wq,
                                const float* __restrict__ bq,
                                const float* __restrict__ Wk,
                                const float* __restrict__ Wv,
                                float* __restrict__ kq,
                                float* __restrict__ vmol,
                                float* __restrict__ cmol) {
    __shared__ float kk[D128];
    int b = blockIdx.x;
    int t = threadIdx.x;           // 128 threads
    float q  = Q[b];
    float e0 = fmaxf(q, 0.0f), e1 = fmaxf(-q, 0.0f);
    float r0 = e0 / fmaxf(e0, 1.0f), r1 = e1 / fmaxf(e1, 1.0f);
    float kd = Wk[t * 2 + 0] * r0 + Wk[t * 2 + 1] * r1;
    float vd = Wv[t * 2 + 0] * e0 + Wv[t * 2 + 1] * e1;
    kk[t] = kd;
    vmol[b * D128 + t] = vd;
    __syncthreads();
    float acc = 0.0f, c = 0.0f;
#pragma unroll 4
    for (int i = 0; i < D128; ++i) {
        float ki = kk[i];
        acc = fmaf(ki, Wq[i * D128 + t], acc);   // Wq^T k  (coalesced over t)
        c   = fmaf(ki, bq[i], c);
    }
    kq[b * D128 + t] = acc;
    if (t == 0) cmol[b] = c;
}

// ------------------------------------------------- weights f32 -> f16 (packed)
__global__ void cvt_w_kernel(const float* __restrict__ W1,
                             const float* __restrict__ W2,
                             const float* __restrict__ Wout,
                             _Float16* __restrict__ o) {
    int i = blockIdx.x * blockDim.x + threadIdx.x;   // 3*16384
    if (i >= 3 * D128 * D128) return;
    const float* src = (i < 16384) ? W1 : (i < 32768 ? W2 : Wout);
    o[i] = (_Float16)src[i & 16383];
}

// ------------------------------------------------- pass 1: dot/softplus/segsum
// dot_n = emb_n . kq[seg] + c[seg];  a = softplus(dot/sqrt(D)); anorm[seg]+=a
__global__ __launch_bounds__(256)
void attn_pass_kernel(const float* __restrict__ emb,
                      const int*   __restrict__ seg,
                      const float* __restrict__ kq,
                      const float* __restrict__ cmol,
                      float* __restrict__ aatom,
                      float* __restrict__ anorm, int N) {
    int lane = threadIdx.x & 31;
    int wave = (blockIdx.x * blockDim.x + threadIdx.x) >> 5;
    int nw   = (gridDim.x * blockDim.x) >> 5;
    for (int n = wave; n < N; n += nw) {
        int s = seg[n];
        float4 ev = *(const float4*)(emb + (size_t)n * D128 + lane * 4);
        float4 kv = *(const float4*)(kq  + (size_t)s * D128 + lane * 4);
        float d = ev.x * kv.x + ev.y * kv.y + ev.z * kv.z + ev.w * kv.w;
#pragma unroll
        for (int off = 16; off > 0; off >>= 1) d += __shfl_xor(d, off, 32);
        if (lane == 0) {
            d = (d + cmol[s]) * 0.08838834764831843f;   // 1/sqrt(128)
            float a = (d > 15.0f) ? d : log1pf(__expf(d));
            aatom[n] = a;
            atomicAdd(&anorm[s], a);
        }
    }
}

// ------------------------------------------------- pass 2: fused MLP via WMMA
// per 16-atom tile: x = scale*v[seg]; y1=silu(x); t=y1@W1.T+b1; y2=silu(t);
// h = x + y2@W2.T + b2; out = silu(h)@Wout.T
__global__ __launch_bounds__(256)
void mlp_tiles_kernel(const float* __restrict__ aatom,
                      const float* __restrict__ anorm,
                      const int*   __restrict__ seg,
                      const float* __restrict__ vmol,
                      const _Float16* __restrict__ Wh,   // W1h|W2h|Wouth row-major
                      const float* __restrict__ b1,
                      const float* __restrict__ b2,
                      float* __restrict__ out, int N) {
    extern __shared__ char smem[];
    _Float16* w1s = (_Float16*)smem;            // 128*TW halves
    _Float16* w2s = w1s + D128 * TW;
    _Float16* w3s = w2s + D128 * TW;
    float* b1s = (float*)(w3s + D128 * TW);
    float* b2s = b1s + D128;
    char*  wavebase = (char*)(b2s + D128);
    const int PER_WAVE = 2 * 16 * TW * 2 + 16 * XW * 4 + 16 * 4 + 16 * 4; // 17280B

    int wid  = threadIdx.x >> 5;
    int lane = threadIdx.x & 31;
    char* my = wavebase + (size_t)wid * PER_WAVE;
    _Float16* A0 = (_Float16*)my;               // 16 x TW f16
    _Float16* A1 = A0 + 16 * TW;
    float* xt  = (float*)(A1 + 16 * TW);        // 16 x XW f32
    float* scl = xt + 16 * XW;                  // 16
    int*   sg  = (int*)(scl + 16);              // 16

    // stage weights into padded LDS (u32 copies, conflict-free)
    const uint32_t* wsrc = (const uint32_t*)Wh;
    for (int idx = threadIdx.x; idx < 3 * 8192; idx += 256) {
        int mtx = idx >> 13;
        int j   = idx & 8191;
        int row = j >> 6, cp = j & 63;
        _Float16* dst = (mtx == 0) ? w1s : (mtx == 1 ? w2s : w3s);
        *(uint32_t*)(dst + row * TW + cp * 2) = wsrc[idx];
    }
    for (int idx = threadIdx.x; idx < D128; idx += 256) { b1s[idx] = b1[idx]; b2s[idx] = b2[idx]; }
    __syncthreads();

    int numTiles = (N + 15) >> 4;
    int gw = gridDim.x * 8;
    int h  = lane >> 4;        // half-wave selector
    int ln = lane & 15;

    for (int tile = blockIdx.x * 8 + wid; tile < numTiles; tile += gw) {
        int n0 = tile << 4;
        if (lane < 16) {
            int atom = n0 + lane;
            float s = 0.0f; int sv = 0;
            if (atom < N) { sv = seg[atom]; s = aatom[atom] / (anorm[sv] + EPS); }
            scl[lane] = s; sg[lane] = sv;
        }
        // build x (f32) and y1 = silu(x) (f16, A-layout source tile)
#pragma unroll 1
        for (int i = lane; i < 512; i += 32) {
            int m = i >> 5, dg = i & 31, d = dg * 4;
            float s = scl[m]; int sv = sg[m];
            float4 vv = *(const float4*)(vmol + (size_t)sv * D128 + d);
            float x0 = s * vv.x, x1 = s * vv.y, x2 = s * vv.z, x3 = s * vv.w;
            *(float4*)(xt + m * XW + d) = make_float4(x0, x1, x2, x3);
            v4h y; y[0] = (_Float16)silu_f(x0); y[1] = (_Float16)silu_f(x1);
                   y[2] = (_Float16)silu_f(x2); y[3] = (_Float16)silu_f(x3);
            *(v4h*)(A0 + m * TW + d) = y;
        }

        v16h af[4];
        // ---------------- GEMM1: t = y1 @ W1.T + b1 ; A1 = silu(t) ----------
#pragma unroll
        for (int kc = 0; kc < 4; ++kc) {
            const _Float16* p = A0 + ln * TW + 32 * kc + 8 * h;
            af[kc] = ld_frag2(p, p + 16);
        }
#pragma unroll 2
        for (int nt = 0; nt < 8; ++nt) {
            v8f acc = {};
            int n = nt * 16 + ln;
#pragma unroll
            for (int kc = 0; kc < 4; ++kc) {
                const _Float16* wb = w1s + n * TW + 32 * kc + 8 * h;
                v16h bf = ld_frag2(wb, wb + 16);
                acc = __builtin_amdgcn_wmma_f32_16x16x32_f16(false, af[kc], false, bf,
                                                             (short)0, acc, false, false);
            }
            float bb = b1s[n];
#pragma unroll
            for (int v = 0; v < 8; ++v) {
                int m = 8 * h + v;
                A1[m * TW + n] = (_Float16)silu_f(acc[v] + bb);
            }
        }
        // ---------------- GEMM2: h = x + y2 @ W2.T + b2 ; A0 = silu(h) ------
#pragma unroll
        for (int kc = 0; kc < 4; ++kc) {
            const _Float16* p = A1 + ln * TW + 32 * kc + 8 * h;
            af[kc] = ld_frag2(p, p + 16);
        }
#pragma unroll 2
        for (int nt = 0; nt < 8; ++nt) {
            v8f acc = {};
            int n = nt * 16 + ln;
#pragma unroll
            for (int kc = 0; kc < 4; ++kc) {
                const _Float16* wb = w2s + n * TW + 32 * kc + 8 * h;
                v16h bf = ld_frag2(wb, wb + 16);
                acc = __builtin_amdgcn_wmma_f32_16x16x32_f16(false, af[kc], false, bf,
                                                             (short)0, acc, false, false);
            }
            float bb = b2s[n];
#pragma unroll
            for (int v = 0; v < 8; ++v) {
                int m = 8 * h + v;
                float hh = xt[m * XW + n] + acc[v] + bb;
                A0[m * TW + n] = (_Float16)silu_f(hh);
            }
        }
        // ---------------- GEMM3: out = y3 @ Wout.T -> xt (f32 staging) ------
#pragma unroll
        for (int kc = 0; kc < 4; ++kc) {
            const _Float16* p = A0 + ln * TW + 32 * kc + 8 * h;
            af[kc] = ld_frag2(p, p + 16);
        }
#pragma unroll 2
        for (int nt = 0; nt < 8; ++nt) {
            v8f acc = {};
            int n = nt * 16 + ln;
#pragma unroll
            for (int kc = 0; kc < 4; ++kc) {
                const _Float16* wb = w3s + n * TW + 32 * kc + 8 * h;
                v16h bf = ld_frag2(wb, wb + 16);
                acc = __builtin_amdgcn_wmma_f32_16x16x32_f16(false, af[kc], false, bf,
                                                             (short)0, acc, false, false);
            }
#pragma unroll
            for (int v = 0; v < 8; ++v) xt[(8 * h + v) * XW + n] = acc[v];
        }
        // coalesced float4 store of the 16x128 output tile
#pragma unroll 1
        for (int i = lane; i < 512; i += 32) {
            int m = i >> 5, dg = i & 31, d = dg * 4;
            int atom = n0 + m;
            if (atom < N)
                *(float4*)(out + (size_t)atom * D128 + d) = *(const float4*)(xt + m * XW + d);
        }
    }
}

// ---------------------------------------------------------------------------
extern "C" void kernel_launch(void* const* d_in, const int* in_sizes, int n_in,
                              void* d_out, int out_size, void* d_ws, size_t ws_size,
                              hipStream_t stream) {
    const float* emb  = (const float*)d_in[0];
    const float* Q    = (const float*)d_in[1];
    const int*   seg  = (const int*)  d_in[2];
    const float* Wq   = (const float*)d_in[3];
    const float* bq   = (const float*)d_in[4];
    const float* Wk   = (const float*)d_in[5];
    const float* Wv   = (const float*)d_in[6];
    const float* W1   = (const float*)d_in[7];
    const float* b1   = (const float*)d_in[8];
    const float* W2   = (const float*)d_in[9];
    const float* b2   = (const float*)d_in[10];
    const float* Wout = (const float*)d_in[11];
    float* out = (float*)d_out;

    int N = in_sizes[0] / D128;
    int B = in_sizes[1];

    // workspace partition
    float* kq    = (float*)d_ws;                    // B*128
    float* vmol  = kq   + (size_t)B * D128;         // B*128
    float* cmol  = vmol + (size_t)B * D128;         // B
    float* anorm = cmol + B;                        // B
    float* aatom = anorm + B;                       // N
    _Float16* Wh = (_Float16*)(aatom + N);          // 3*16384 halves

    zero_f32_kernel<<<(B + 255) / 256, 256, 0, stream>>>(anorm, B);
    mol_prep_kernel<<<B, D128, 0, stream>>>(Q, Wq, bq, Wk, Wv, kq, vmol, cmol);
    cvt_w_kernel<<<(3 * D128 * D128 + 255) / 256, 256, 0, stream>>>(W1, W2, Wout, Wh);
    attn_pass_kernel<<<2048, 256, 0, stream>>>(emb, seg, kq, cmol, aatom, anorm, N);

    size_t smem = (size_t)3 * D128 * TW * 2      // weights f16
                + 2 * D128 * 4                   // biases f32
                + 8 * (2 * 16 * TW * 2 + 16 * XW * 4 + 16 * 4 + 16 * 4); // per-wave
    mlp_tiles_kernel<<<1024, 256, smem, stream>>>(aatom, anorm, seg, vmol, Wh,
                                                  b1, b2, out, N);
}
